// ENN_20203526160942
// MI455X (gfx1250) — matrix operand order repaired and checked by reading, not compiled, not run on hardware
//
#include <hip/hip_runtime.h>
#include <hip/hip_bf16.h>
#include <math.h>

typedef __attribute__((ext_vector_type(16))) _Float16 v16h;
typedef __attribute__((ext_vector_type(8)))  float    v8f;

#define N_ATOMS 1024
#define L       64
#define TB      16
#define NBLK    (N_ATOMS / TB)   // 64

// Native gfx1250 v_tanh_f32 (single TRANS op, co-executes with WMMA);
// guarded fallback keeps both toolchains + host pass compiling.
__device__ __forceinline__ float fast_tanh(float x) {
#if __has_builtin(__builtin_amdgcn_tanhf)
    return __builtin_amdgcn_tanhf(x);
#else
    return tanhf(x);
#endif
}

// ---------------------------------------------------------------------------
// k1: H = learnable_matrix[cats]; pre_i = H @ We1[0:64]; pre_j = H @ We1[64:128]
// ---------------------------------------------------------------------------
__global__ __launch_bounds__(256) void k1_embed(
    const int* __restrict__ cats, const float* __restrict__ Lm,
    const float* __restrict__ We1, float* __restrict__ H,
    float* __restrict__ prei, float* __restrict__ prej)
{
    __shared__ __align__(16) float Hs[TB][L];
    const int t  = threadIdx.x;
    const int ib = blockIdx.x;

    #pragma unroll
    for (int q = 0; q < 4; ++q) {
        int e = t + 256 * q;
        int r = e >> 6, c = e & 63;
        float h = Lm[cats[ib * TB + r] * L + c];
        Hs[r][c] = h;
        H[(ib * TB + r) * L + c] = h;
    }
    __syncthreads();

    const int c  = t & 63;
    const int r0 = t >> 6;                 // 0..3
    for (int rr = r0; rr < TB; rr += 4) {
        float si = 0.f, sj = 0.f;
        #pragma unroll 8
        for (int k = 0; k < L; ++k) {
            float h = Hs[rr][k];
            si += h * We1[k * L + c];
            sj += h * We1[(L + k) * L + c];
        }
        prei[(ib * TB + rr) * L + c] = si;
        prej[(ib * TB + rr) * L + c] = sj;
    }
}

// Gather the 16 floats one lane needs for an A fragment K-half:
// indices base0..base0+7 and base0+16..base0+23 (base0 is 32B aligned).
__device__ __forceinline__ void load16(const float* __restrict__ row, int base0,
                                       float out[16])
{
    const float4* v = (const float4*)(row + base0);
    float4 a0 = v[0], a1 = v[1], a2 = v[4], a3 = v[5];
    out[0] = a0.x;  out[1] = a0.y;  out[2]  = a0.z;  out[3]  = a0.w;
    out[4] = a1.x;  out[5] = a1.y;  out[6]  = a1.z;  out[7]  = a1.w;
    out[8] = a2.x;  out[9] = a2.y;  out[10] = a2.z;  out[11] = a2.w;
    out[12] = a3.x; out[13] = a3.y; out[14] = a3.z;  out[15] = a3.w;
}

// ---------------------------------------------------------------------------
// k2: fused edge MLP. One block per (ib, jb) 16x16 atom-pair tile.
//   h1[p,k] = tanh(pre_i[ii,k] + pre_j[jj,k] + d2(ii,jj)*w1z[k] + be1[k])
//   s[ii,jj] = sum_n tanh( (h1 @ We2)[p,n] + be2[n] )     (p = ii*16+jj)
// GEMM M=256,K=64,N=64 via v_wmma_f32_16x16x32_f16; M-tile index == ii.
// ---------------------------------------------------------------------------
__global__ __launch_bounds__(256) void k2_edge(
    const float* __restrict__ prei, const float* __restrict__ prej,
    const float* __restrict__ Z,    const float* __restrict__ We1,
    const float* __restrict__ be1,  const float* __restrict__ We2,
    const float* __restrict__ be2,  float* __restrict__ Spart)
{
    __shared__ __align__(16) float    ci[TB][L];      // pre_i + be1
    __shared__ __align__(16) float    cj[TB][L];      // pre_j
    __shared__ __align__(16) float    wz[L];          // We1 row 2L
    __shared__ __align__(16) float    b2[L];          // be2
    __shared__ __align__(16) float    d2t[TB * TB];
    __shared__ __align__(16) _Float16 w2hT[L * L];    // We2 transposed: [n][k]
    __shared__ __align__(16) float    stile[TB * TB];
    __shared__ __align__(16) float    zi[TB][4], zj[TB][4];

    const int t  = threadIdx.x;
    const int ib = blockIdx.x >> 6;       // NBLK == 64
    const int jb = blockIdx.x & 63;

    // ---- stage small operands -------------------------------------------
    if (t < 48)       { int r = t / 3, c = t % 3;           zi[r][c] = Z[(ib * TB + r) * 3 + c]; }
    else if (t < 96)  { int u = t - 48, r = u / 3, c = u % 3; zj[r][c] = Z[(jb * TB + r) * 3 + c]; }
    else if (t < 160) { wz[t - 96]  = We1[2 * L * L + (t - 96)]; }
    else if (t < 224) { b2[t - 160] = be2[t - 160]; }

    #pragma unroll
    for (int q = 0; q < 4; ++q) {
        int e = t + 256 * q;
        int r = e >> 6, c = e & 63;
        ci[r][c] = prei[(ib * TB + r) * L + c] + be1[c];
        cj[r][c] = prej[(jb * TB + r) * L + c];
    }
    // Stage We2 transposed as f16: w2hT[n*L + k] = We2[k*L + n]
    #pragma unroll
    for (int q = 0; q < 16; ++q) {
        int e = t + 256 * q;
        int k = e >> 6, n = e & 63;
        w2hT[n * L + k] = (_Float16)We2[k * L + n];
    }
    __syncthreads();

    // ---- d2 tile ---------------------------------------------------------
    {
        int ii = t >> 4, jj = t & 15;
        float dx = zi[ii][0] - zj[jj][0];
        float dy = zi[ii][1] - zj[jj][1];
        float dz = zi[ii][2] - zj[jj][2];
        d2t[t] = dx * dx + dy * dy + dz * dz;
    }
    __syncthreads();

    // ---- WMMA GEMM -------------------------------------------------------
    const int wave    = t >> 5;           // 0..7 -> M-tiles 2w, 2w+1
    const int lane    = t & 31;
    const int halfsel = lane >> 4;        // K-half selector
    const int lid     = lane & 15;

    // Preload B fragments. ISA B layout: lane = n + 16*khalf16;
    // elem e -> K = ks*32 + e + 8*halfsel (+8 if e>=8). Transposed LDS makes the 16
    // halves contiguous -> merges into ds_load_b128 pairs.
    v16h bfr[2][4];
    #pragma unroll
    for (int ks = 0; ks < 2; ++ks)
        #pragma unroll
        for (int nt = 0; nt < 4; ++nt) {
            const int n = nt * 16 + lid;
            const _Float16* bp = &w2hT[n * L + ks * 32 + 8 * halfsel];
            v16h b;
            #pragma unroll
            for (int e = 0; e < 16; ++e) b[e] = bp[e + (e & 8)];
            bfr[ks][nt] = b;
        }

    #pragma unroll
    for (int mi = 0; mi < 2; ++mi) {
        const int mt = wave * 2 + mi;     // == ii (local i index)
        v8f acc0 = {}, acc1 = {}, acc2 = {}, acc3 = {};
        const float d2v = d2t[mt * TB + lid];   // row within M-tile == jj == lid

        #pragma unroll
        for (int ks = 0; ks < 2; ++ks) {
            // A fragment, ISA layout: lane = m + 16*h;
            // elem e -> K = ks*32 + e + 8h (+8 if e>=8). Two 8-float runs.
            const int base0 = ks * 32 + 8 * halfsel;
            float xi[16], xj[16], xw[16];
            load16(&ci[mt][0], base0, xi);
            load16(&cj[lid][0], base0, xj);
            load16(&wz[0],      base0, xw);
            v16h a;
            #pragma unroll
            for (int e = 0; e < 16; ++e)
                a[e] = (_Float16)fast_tanh(fmaf(d2v, xw[e], xi[e] + xj[e]));

            acc0 = __builtin_amdgcn_wmma_f32_16x16x32_f16(false, a, false, bfr[ks][0], (short)0, acc0, false, false);
            acc1 = __builtin_amdgcn_wmma_f32_16x16x32_f16(false, a, false, bfr[ks][1], (short)0, acc1, false, false);
            acc2 = __builtin_amdgcn_wmma_f32_16x16x32_f16(false, a, false, bfr[ks][2], (short)0, acc2, false, false);
            acc3 = __builtin_amdgcn_wmma_f32_16x16x32_f16(false, a, false, bfr[ks][3], (short)0, acc3, false, false);
        }

        // ---- epilogue: s[jj] = sum_n tanh(acc + be2[n]) ------------------
        // C layout: VGPR v, lane -> row (jj) = v + 8*halfsel, col = lid (+16*nt)
        float rowsum[8];
        #pragma unroll
        for (int v = 0; v < 8; ++v) {
            float s = fast_tanh(acc0[v] + b2[ 0 + lid])
                    + fast_tanh(acc1[v] + b2[16 + lid])
                    + fast_tanh(acc2[v] + b2[32 + lid])
                    + fast_tanh(acc3[v] + b2[48 + lid]);
            #pragma unroll
            for (int off = 1; off < 16; off <<= 1)    // stays within 16-lane half
                s += __shfl_xor(s, off, 32);
            rowsum[v] = s;
        }
        if (lid == 0) {                               // lanes 0 and 16
            #pragma unroll
            for (int v = 0; v < 8; ++v)
                stile[mt * TB + (v + 8 * halfsel)] = rowsum[v];
        }
    }
    __syncthreads();

    // ---- per-i tile totals (diag removed when ib==jb) --------------------
    if (t < TB) {
        float tot = 0.f;
        #pragma unroll
        for (int jj = 0; jj < TB; ++jj) tot += stile[t * TB + jj];
        if (ib == jb) tot -= stile[t * TB + t];
        Spart[(jb * NBLK + ib) * TB + t] = tot;       // deterministic, no atomics
    }
}

// ---------------------------------------------------------------------------
// k3: V[i] = sum_jb Spart[jb][i];  hh = tanh(H@Wh1[:L] + V*Wh1[L] + bh1);
//     newH = tanh(hh@Wh2 + bh2);  per-block column sums -> Hpart[ib][c]
// ---------------------------------------------------------------------------
__global__ __launch_bounds__(256) void k3_node(
    const float* __restrict__ H,   const float* __restrict__ Spart,
    const float* __restrict__ Wh1, const float* __restrict__ bh1,
    const float* __restrict__ Wh2, const float* __restrict__ bh2,
    float* __restrict__ Hpart)
{
    __shared__ __align__(16) float Hs[TB][L];
    __shared__ __align__(16) float hhs[TB][L];
    __shared__ float Vs[TB];
    __shared__ float colred[4][L];

    const int t  = threadIdx.x;
    const int ib = blockIdx.x;

    #pragma unroll
    for (int q = 0; q < 4; ++q) {
        int e = t + 256 * q;
        int r = e >> 6, c = e & 63;
        Hs[r][c] = H[(ib * TB + r) * L + c];
    }
    if (t < TB) {
        float v = 0.f;
        for (int jb = 0; jb < NBLK; ++jb)
            v += Spart[(jb * NBLK + ib) * TB + t];    // fixed order -> deterministic
        Vs[t] = v;
    }
    __syncthreads();

    const int c  = t & 63;
    const int r0 = t >> 6;                            // 0..3
    for (int rr = r0; rr < TB; rr += 4) {
        float s = bh1[c] + Vs[rr] * Wh1[L * L + c];
        #pragma unroll 8
        for (int k = 0; k < L; ++k) s += Hs[rr][k] * Wh1[k * L + c];
        hhs[rr][c] = fast_tanh(s);
    }
    __syncthreads();

    float part = 0.f;
    for (int rr = r0; rr < TB; rr += 4) {
        float s = bh2[c];
        #pragma unroll 8
        for (int k = 0; k < L; ++k) s += hhs[rr][k] * Wh2[k * L + c];
        part += fast_tanh(s);
    }
    colred[r0][c] = part;
    __syncthreads();
    if (r0 == 0)
        Hpart[ib * L + c] = colred[0][c] + colred[1][c] + colred[2][c] + colred[3][c];
}

// ---------------------------------------------------------------------------
// k4: out = (sum_ib Hpart[ib]) @ Wo + bo
// ---------------------------------------------------------------------------
__global__ void k4_out(const float* __restrict__ Hpart, const float* __restrict__ Wo,
                       const float* __restrict__ bo, float* __restrict__ out)
{
    __shared__ float red[L];
    const int t = threadIdx.x;                        // 64 threads
    float s = 0.f;
    for (int b = 0; b < NBLK; ++b) s += Hpart[b * L + t];
    red[t] = s * Wo[t];
    __syncthreads();
    if (t == 0) {
        float acc = bo[0];
        for (int i = 0; i < L; ++i) acc += red[i];
        out[0] = acc;
    }
}

// ---------------------------------------------------------------------------
extern "C" void kernel_launch(void* const* d_in, const int* in_sizes, int n_in,
                              void* d_out, int out_size, void* d_ws, size_t ws_size,
                              hipStream_t stream)
{
    const int*   cats = (const int*)  d_in[0];
    const float* Z    = (const float*)d_in[1];
    const float* Lm   = (const float*)d_in[2];
    const float* We1  = (const float*)d_in[3];
    const float* be1  = (const float*)d_in[4];
    const float* We2  = (const float*)d_in[5];
    const float* be2  = (const float*)d_in[6];
    const float* Wh1  = (const float*)d_in[7];
    const float* bh1  = (const float*)d_in[8];
    const float* Wh2  = (const float*)d_in[9];
    const float* bh2  = (const float*)d_in[10];
    const float* Wo   = (const float*)d_in[11];
    const float* bo   = (const float*)d_in[12];
    float* out = (float*)d_out;

    // Workspace layout (floats): H | pre_i | pre_j | Spart | Hpart  (~1.04 MB)
    float* ws    = (float*)d_ws;
    float* H     = ws;
    float* prei  = H     + N_ATOMS * L;
    float* prej  = prei  + N_ATOMS * L;
    float* Spart = prej  + N_ATOMS * L;
    float* Hpart = Spart + NBLK * NBLK * TB;

    k1_embed<<<NBLK,        256, 0, stream>>>(cats, Lm, We1, H, prei, prej);
    k2_edge <<<NBLK * NBLK, 256, 0, stream>>>(prei, prej, Z, We1, be1, We2, be2, Spart);
    k3_node <<<NBLK,        256, 0, stream>>>(H, Spart, Wh1, bh1, Wh2, bh2, Hpart);
    k4_out  <<<1,            64, 0, stream>>>(Hpart, Wo, bo, out);
}
